// MemoryGaussianMoELayer_67491116089919
// MI455X (gfx1250) — matrix operand order repaired and checked by reading, not compiled, not run on hardware
//
#include <hip/hip_runtime.h>
#include <hip/hip_bf16.h>
#include <math.h>

// ---------------- problem dims ----------------
#define B_DIM   2
#define T_DIM   2048
#define NTOK    4096      // B*T
#define E_DIM   512
#define HID_DIM 2048
#define OUT_DIM 512
#define NE_     8
#define TOPK_   2
#define NH_     4
#define DH_     128       // E/NH
#define M_HIST  256

typedef __bf16 bf16;
typedef __attribute__((ext_vector_type(16))) __bf16 bf16x16;
typedef __attribute__((ext_vector_type(8)))  __bf16 bf16x8;
typedef __attribute__((ext_vector_type(8)))  float  f32x8;

// ==========================================================================
// WMMA GEMM:  C[M][N] = A[M][K] (bf16, row-major) * Bt[N][K]^T (bf16, N-major)
// Each wave32 computes a 32x64 strip: 2 A-fragments x 4 B-fragments
// -> 8 v_wmma per K-step against 12 b128 loads (1.5 loads/WMMA) while
// keeping 64 accumulator VGPRs (no spills; 32x128 spilled to scratch).
// block = (32, 8) -> 8 waves cover 512 columns; grid = (ceil(N/512), M/32).
// ==========================================================================
#define GM_F32       0
#define GM_BF16      1
#define GM_GELU_BF16 2
#define GM_MOE_ACC   3

__device__ __forceinline__ bf16x16 load_a_frag(const bf16* A, int lda, int row,
                                               int kk, int lane) {
    // 16-bit A 16x32 layout: lane<16 holds K = kb..kb+7 and kb+16..kb+23 (kb=0)
    //                        lane>=16: kb=8
    const int kb = kk + ((lane >> 4) << 3);
    const bf16* p = A + (size_t)row * lda + kb;
    bf16x8 lo = *(const bf16x8*)(p);
    bf16x8 hi = *(const bf16x8*)(p + 16);
    bf16x16 r;
#pragma unroll
    for (int j = 0; j < 8; ++j) { r[j] = lo[j]; r[j + 8] = hi[j]; }
    return r;
}

__device__ __forceinline__ bf16x16 load_b_frag(const bf16* Bt, int ldb, int col,
                                               int kk, int lane) {
    // 16-bit B 32x16 layout: lane&15 = N, lanes 0-15 hold K=kk..kk+7 and kk+16..kk+23,
    //                        lanes 16-31 hold K=kk+8..kk+15 and kk+24..kk+31
    const int kb = kk + ((lane >> 4) << 3);
    const bf16* p = Bt + (size_t)col * ldb + kb;
    bf16x8 lo = *(const bf16x8*)(p);
    bf16x8 hi = *(const bf16x8*)(p + 16);
    bf16x16 r;
#pragma unroll
    for (int j = 0; j < 8; ++j) { r[j] = lo[j]; r[j + 8] = hi[j]; }
    return r;
}

__global__ void __launch_bounds__(256)
wmma_gemm_kernel(const bf16* __restrict__ A,
                 const bf16* __restrict__ Bt,
                 const float* __restrict__ bias,
                 float* __restrict__ Cf,
                 bf16*  __restrict__ Cb,
                 const float* __restrict__ gates, int gstride,
                 int Mrows, int Ncols, int K, int mode) {
    const int lane = threadIdx.x;                 // 0..31
    const int wave = threadIdx.y;                 // 0..7
    const int tn = (blockIdx.x * 8 + wave) * 64;  // first column of strip
    const int tm = blockIdx.y * 32;               // first row of strip
    if (tn >= Ncols || tm >= Mrows) return;       // uniform per wave

    f32x8 acc[2][4];
#pragma unroll
    for (int mi = 0; mi < 2; ++mi)
#pragma unroll
        for (int s = 0; s < 4; ++s)
#pragma unroll
            for (int r = 0; r < 8; ++r) acc[mi][s][r] = 0.0f;

    const int arow0 = tm + (lane & 15);
    for (int kk = 0; kk < K; kk += 32) {
        // prefetch A-stream ahead (global_prefetch_b8)
        __builtin_prefetch(A + (size_t)arow0 * K + kk + 64, 0, 3);
        __builtin_prefetch(A + (size_t)(arow0 + 16) * K + kk + 64, 0, 3);
        bf16x16 a0 = load_a_frag(A, K, arow0,      kk, lane);
        bf16x16 a1 = load_a_frag(A, K, arow0 + 16, kk, lane);
#pragma unroll
        for (int s = 0; s < 4; ++s) {
            bf16x16 b = load_b_frag(Bt, K, tn + s * 16 + (lane & 15), kk, lane);
            acc[0][s] = __builtin_amdgcn_wmma_f32_16x16x32_bf16(
                false, a0, false, b, (short)0, acc[0][s], false, false);
            acc[1][s] = __builtin_amdgcn_wmma_f32_16x16x32_bf16(
                false, a1, false, b, (short)0, acc[1][s], false, false);
        }
    }

    // C layout: element r -> row = tm + mi*16 + 8*(lane>>4) + r, col = lane&15
    const int rowoff = (lane >> 4) << 3;
    const int c0 = lane & 15;
#pragma unroll
    for (int mi = 0; mi < 2; ++mi) {
#pragma unroll
        for (int r = 0; r < 8; ++r) {
            const int row = tm + mi * 16 + rowoff + r;
            const float g =
                (mode == GM_MOE_ACC) ? gates[(size_t)row * gstride] : 0.f;
#pragma unroll
            for (int s = 0; s < 4; ++s) {
                const int col = tn + s * 16 + c0;
                float v = acc[mi][s][r] + (bias ? bias[col] : 0.0f);
                const size_t off = (size_t)row * Ncols + col;
                if (mode == GM_F32) {
                    Cf[off] = v;
                } else if (mode == GM_BF16) {
                    Cb[off] = (bf16)v;
                } else if (mode == GM_GELU_BF16) {
                    v = 0.5f * v * (1.0f + erff(v * 0.7071067811865476f));
                    Cb[off] = (bf16)v;
                } else { // GM_MOE_ACC
                    Cf[off] += g * v;
                }
            }
        }
    }
}

// ==========================================================================
// f32 -> bf16 convert (no transpose)
// ==========================================================================
__global__ void conv_bf16_kernel(const float* __restrict__ src,
                                 bf16* __restrict__ dst, int n) {
    for (int i = blockIdx.x * blockDim.x + threadIdx.x; i < n;
         i += gridDim.x * blockDim.x)
        dst[i] = (bf16)src[i];
}

// ==========================================================================
// Tiled transpose + convert: src f32 [rows][ld] -> dst bf16 [cols][rows]
// (dst[n][k] = src[k][n]); block (32,8), 32x32 tiles through LDS
// ==========================================================================
__global__ void transpose_bf16_kernel(const float* __restrict__ src, int ld,
                                      int rows, int cols,
                                      bf16* __restrict__ dst) {
    __shared__ float tile[32][33];
    const int c0 = blockIdx.x * 32, r0 = blockIdx.y * 32;
#pragma unroll
    for (int i = 0; i < 4; ++i) {
        int r = r0 + threadIdx.y + i * 8, c = c0 + threadIdx.x;
        if (r < rows && c < cols)
            tile[threadIdx.y + i * 8][threadIdx.x] = src[(size_t)r * ld + c];
    }
    __syncthreads();
#pragma unroll
    for (int i = 0; i < 4; ++i) {
        int c = c0 + threadIdx.y + i * 8, r = r0 + threadIdx.x;
        if (c < cols && r < rows)
            dst[(size_t)c * rows + r] = (bf16)tile[threadIdx.x][threadIdx.y + i * 8];
    }
}

// ==========================================================================
// Attention: one block (256 thr) per token. q2[N][E] f32, k2/v2[M][E] f32
// heads = 4, dh = 128, M = 256.  ctx written as bf16 [N][E].
// ==========================================================================
__global__ void attention_kernel(const float* __restrict__ q2,
                                 const float* __restrict__ k2,
                                 const float* __restrict__ v2,
                                 bf16* __restrict__ ctx_bf) {
    __shared__ float qs[E_DIM];
    __shared__ float sc[NH_ * M_HIST];
    __shared__ float red[256];
    const int tok = blockIdx.x, tid = threadIdx.x;

    qs[tid]       = q2[(size_t)tok * E_DIM + tid];
    qs[tid + 256] = q2[(size_t)tok * E_DIM + tid + 256];
    __syncthreads();

    // scores: thread tid handles key m = tid for all 4 heads
    const float scale = 0.08838834764831845f; // 1/sqrt(128)
    const int m = tid;
#pragma unroll
    for (int h = 0; h < NH_; ++h) {
        float d = 0.f;
        const float* kp = k2 + (size_t)m * E_DIM + h * DH_;
        const float* qp = qs + h * DH_;
        for (int i = 0; i < DH_; ++i) d += qp[i] * kp[i];
        sc[h * M_HIST + m] = d * scale;
    }
    __syncthreads();

    // softmax per head: head g handled by 64 threads
    const int g = tid >> 6, t = tid & 63;
    float lm = -1e30f;
#pragma unroll
    for (int j = 0; j < 4; ++j) lm = fmaxf(lm, sc[g * M_HIST + t + j * 64]);
    red[tid] = lm; __syncthreads();
    for (int off = 32; off > 0; off >>= 1) {
        if (t < off) red[tid] = fmaxf(red[tid], red[tid + off]);
        __syncthreads();
    }
    const float hmax = red[g << 6];
    __syncthreads();
    float ls = 0.f;
#pragma unroll
    for (int j = 0; j < 4; ++j) {
        float e = __expf(sc[g * M_HIST + t + j * 64] - hmax);
        sc[g * M_HIST + t + j * 64] = e; ls += e;
    }
    red[tid] = ls; __syncthreads();
    for (int off = 32; off > 0; off >>= 1) {
        if (t < off) red[tid] += red[tid + off];
        __syncthreads();
    }
    const float inv = 1.0f / red[g << 6];
    __syncthreads();
#pragma unroll
    for (int j = 0; j < 4; ++j) sc[g * M_HIST + t + j * 64] *= inv;
    __syncthreads();

    // ctx: 512 outputs, 2 per thread
#pragma unroll
    for (int i = 0; i < 2; ++i) {
        const int o = tid + i * 256, h = o >> 7, d = o & 127;
        float s = 0.f;
        for (int mm = 0; mm < M_HIST; ++mm)
            s += sc[h * M_HIST + mm] * v2[(size_t)mm * E_DIM + h * DH_ + d];
        ctx_bf[(size_t)tok * E_DIM + o] = (bf16)s;
    }
}

// ==========================================================================
// Fused LayerNorm + gating head: one block (256 thr) per token.
// Writes weights / log_probs / top_i outputs and gates[N][NE] to workspace.
// ==========================================================================
__global__ void ln_gating_kernel(const float* __restrict__ ao,
                                 const float* __restrict__ xf,
                                 const float* __restrict__ ln_g,
                                 const float* __restrict__ ln_b,
                                 const float* __restrict__ Wo,
                                 const float* __restrict__ bo,
                                 const float* __restrict__ Wg,
                                 const float* __restrict__ bg,
                                 float* __restrict__ gates,
                                 float* __restrict__ out_logp,
                                 float* __restrict__ out_w,
                                 float* __restrict__ out_topi) {
    __shared__ float lns[E_DIM];
    __shared__ float red[256];
    __shared__ float lg[16];
    const int tok = blockIdx.x, tid = threadIdx.x;

    const float a0 = ao[(size_t)tok * E_DIM + tid];
    const float a1 = ao[(size_t)tok * E_DIM + tid + 256];
    red[tid] = a0 + a1; __syncthreads();
    for (int off = 128; off > 0; off >>= 1) {
        if (tid < off) red[tid] += red[tid + off];
        __syncthreads();
    }
    const float mu = red[0] * (1.0f / E_DIM);
    __syncthreads();
    const float d0 = a0 - mu, d1 = a1 - mu;
    red[tid] = d0 * d0 + d1 * d1; __syncthreads();
    for (int off = 128; off > 0; off >>= 1) {
        if (tid < off) red[tid] += red[tid + off];
        __syncthreads();
    }
    const float rstd = rsqrtf(red[0] * (1.0f / E_DIM) + 1e-5f);
    __syncthreads();
    lns[tid]       = d0 * rstd * ln_g[tid] + ln_b[tid];
    lns[tid + 256] = d1 * rstd * ln_g[tid + 256] + ln_b[tid + 256];
    __syncthreads();

    // 16 dot products of length 512: 16 groups x 16 threads
    const int j = tid >> 4, s = tid & 15;
    float p = 0.f;
    if (j < NE_) {
        for (int k = s; k < E_DIM; k += 16) p += lns[k] * Wo[k * NE_ + j];
    } else {
        const int jj = j - NE_;
        for (int k = s; k < E_DIM; k += 16)
            p += xf[(size_t)tok * E_DIM + k] * Wg[k * NE_ + jj];
    }
    red[tid] = p; __syncthreads();
    if (s == 0) {
        float tsum = 0.f;
        for (int i = 0; i < 16; ++i) tsum += red[(j << 4) + i];
        lg[j] = tsum + ((j < NE_) ? bo[j] : bg[j - NE_]);
    }
    __syncthreads();

    if (tid == 0) {
        float gl[NE_], w[NE_];
        float mx = -1e30f;
        for (int e = 0; e < NE_; ++e) { gl[e] = lg[e] + lg[NE_ + e]; mx = fmaxf(mx, gl[e]); }
        float sum = 0.f;
        for (int e = 0; e < NE_; ++e) { w[e] = __expf(gl[e] - mx); sum += w[e]; }
        const float inv = 1.0f / sum, lse = __logf(sum);
        for (int e = 0; e < NE_; ++e) {
            out_w[(size_t)tok * NE_ + e]    = w[e] * inv;
            out_logp[(size_t)tok * NE_ + e] = gl[e] - mx - lse;
        }
        int i1 = 0;
        for (int e = 1; e < NE_; ++e) if (w[e] > w[i1]) i1 = e;
        int i2 = (i1 == 0) ? 1 : 0;
        for (int e = 0; e < NE_; ++e) if (e != i1 && w[e] > w[i2]) i2 = e;
        const float w1 = w[i1] * inv, w2 = w[i2] * inv;
        const float dn = 1.0f / (w1 + w2 + 1e-9f);
        for (int e = 0; e < NE_; ++e) gates[(size_t)tok * NE_ + e] = 0.f;
        gates[(size_t)tok * NE_ + i1] = w1 * dn;
        gates[(size_t)tok * NE_ + i2] = w2 * dn;
        out_topi[(size_t)tok * TOPK_ + 0] = (float)i1;
        out_topi[(size_t)tok * TOPK_ + 1] = (float)i2;
    }
}

__global__ void zero_kernel(float* __restrict__ p, int n) {
    for (int i = blockIdx.x * blockDim.x + threadIdx.x; i < n;
         i += gridDim.x * blockDim.x)
        p[i] = 0.f;
}

// ==========================================================================
// host side
// ==========================================================================
static inline void launch_gemm(const bf16* A, const bf16* Bt, const float* bias,
                               float* Cf, bf16* Cb, const float* gates,
                               int gstride, int M, int N, int K, int mode,
                               hipStream_t s) {
    dim3 block(32, 8);
    dim3 grid((N + 511) / 512, M / 32);
    wmma_gemm_kernel<<<grid, block, 0, s>>>(A, Bt, bias, Cf, Cb, gates,
                                            gstride, M, N, K, mode);
}

static inline void launch_tr(const float* src, int ld, int rows, int cols,
                             bf16* dst, hipStream_t s) {
    dim3 block(32, 8);
    dim3 grid((cols + 31) / 32, (rows + 31) / 32);
    transpose_bf16_kernel<<<grid, block, 0, s>>>(src, ld, rows, cols, dst);
}

extern "C" void kernel_launch(void* const* d_in, const int* in_sizes, int n_in,
                              void* d_out, int out_size, void* d_ws, size_t ws_size,
                              hipStream_t stream) {
    const float* x      = (const float*)d_in[0];
    const float* hq     = (const float*)d_in[1];
    const float* hmk    = (const float*)d_in[2];
    const float* Wg     = (const float*)d_in[3];
    const float* bg     = (const float*)d_in[4];
    const float* Wq     = (const float*)d_in[5];
    const float* bq     = (const float*)d_in[6];
    const float* Wk     = (const float*)d_in[7];
    const float* bk     = (const float*)d_in[8];
    const float* Wv     = (const float*)d_in[9];
    const float* bv     = (const float*)d_in[10];
    const float* ipw    = (const float*)d_in[11];   // (E, 3E)
    const float* ipb    = (const float*)d_in[12];   // (3E)
    const float* aow    = (const float*)d_in[13];
    const float* aob    = (const float*)d_in[14];
    const float* ln_g   = (const float*)d_in[15];
    const float* ln_b   = (const float*)d_in[16];
    const float* Wo     = (const float*)d_in[17];
    const float* bo     = (const float*)d_in[18];
    const float* W1     = (const float*)d_in[19];   // (NE,E,HID)
    const float* b1     = (const float*)d_in[20];
    const float* W2     = (const float*)d_in[21];   // (NE,HID,OUT)
    const float* b2     = (const float*)d_in[22];

    float* out_final = (float*)d_out;                          // [N,OUT]
    float* out_logp  = out_final + (size_t)NTOK * OUT_DIM;     // [N,NE]
    float* out_w     = out_logp + (size_t)NTOK * NE_;          // [N,NE]
    float* out_topi  = out_w + (size_t)NTOK * NE_;             // [N,TOPK]

    // ---- workspace carve-up ----
    char* base = (char*)d_ws;
    size_t cur = 0;
    auto carve = [&](size_t bytes) -> char* {
        char* p = base + cur;
        cur = (cur + bytes + 255) & ~(size_t)255;
        return p;
    };
    bf16* x_bf    = (bf16*)carve((size_t)NTOK * E_DIM * 2);
    bf16* hq_bf   = (bf16*)carve((size_t)M_HIST * E_DIM * 2);
    bf16* hmk_bf  = (bf16*)carve((size_t)M_HIST * E_DIM * 2);
    bf16* WqT     = (bf16*)carve((size_t)E_DIM * E_DIM * 2);
    bf16* WkT     = (bf16*)carve((size_t)E_DIM * E_DIM * 2);
    bf16* WvT     = (bf16*)carve((size_t)E_DIM * E_DIM * 2);
    bf16* WqiT    = (bf16*)carve((size_t)E_DIM * E_DIM * 2);
    bf16* WkiT    = (bf16*)carve((size_t)E_DIM * E_DIM * 2);
    bf16* WviT    = (bf16*)carve((size_t)E_DIM * E_DIM * 2);
    bf16* aowT    = (bf16*)carve((size_t)E_DIM * E_DIM * 2);
    bf16* W1T     = (bf16*)carve((size_t)NE_ * HID_DIM * E_DIM * 2);
    bf16* W2T     = (bf16*)carve((size_t)NE_ * OUT_DIM * HID_DIM * 2);
    bf16* q_bf    = (bf16*)carve((size_t)NTOK * E_DIM * 2);
    float* q2_f   = (float*)carve((size_t)NTOK * E_DIM * 4);
    bf16* kf_bf   = (bf16*)carve((size_t)M_HIST * E_DIM * 2);
    bf16* vf_bf   = (bf16*)carve((size_t)M_HIST * E_DIM * 2);
    float* k2_f   = (float*)carve((size_t)M_HIST * E_DIM * 4);
    float* v2_f   = (float*)carve((size_t)M_HIST * E_DIM * 4);
    bf16* ctx_bf  = (bf16*)carve((size_t)NTOK * E_DIM * 2);
    float* ao_f   = (float*)carve((size_t)NTOK * E_DIM * 4);
    float* gates  = (float*)carve((size_t)NTOK * NE_ * 4);
    bf16* h_bf    = (bf16*)carve((size_t)NTOK * HID_DIM * 2);
    (void)ws_size; (void)n_in; (void)in_sizes; (void)out_size;

    // ---- stage 0: converts + transposes to bf16 ----
    conv_bf16_kernel<<<512, 256, 0, stream>>>(x,   x_bf,  NTOK * E_DIM);
    conv_bf16_kernel<<<64,  256, 0, stream>>>(hq,  hq_bf, M_HIST * E_DIM);
    conv_bf16_kernel<<<64,  256, 0, stream>>>(hmk, hmk_bf, M_HIST * E_DIM);
    launch_tr(Wq,  E_DIM,     E_DIM, E_DIM, WqT,  stream);
    launch_tr(Wk,  E_DIM,     E_DIM, E_DIM, WkT,  stream);
    launch_tr(Wv,  E_DIM,     E_DIM, E_DIM, WvT,  stream);
    launch_tr(ipw,           3*E_DIM, E_DIM, E_DIM, WqiT, stream);
    launch_tr(ipw + E_DIM,   3*E_DIM, E_DIM, E_DIM, WkiT, stream);
    launch_tr(ipw + 2*E_DIM, 3*E_DIM, E_DIM, E_DIM, WviT, stream);
    launch_tr(aow, E_DIM,     E_DIM, E_DIM, aowT, stream);
    for (int e = 0; e < NE_; ++e) {
        launch_tr(W1 + (size_t)e * E_DIM * HID_DIM, HID_DIM, E_DIM, HID_DIM,
                  W1T + (size_t)e * HID_DIM * E_DIM, stream);
        launch_tr(W2 + (size_t)e * HID_DIM * OUT_DIM, OUT_DIM, HID_DIM, OUT_DIM,
                  W2T + (size_t)e * OUT_DIM * HID_DIM, stream);
    }

    // ---- stage 1: q/k/v projection chains (WMMA) ----
    launch_gemm(x_bf,  WqT,  bq, nullptr, q_bf,  nullptr, 0, NTOK,   E_DIM, E_DIM, GM_BF16, stream);
    launch_gemm(hq_bf, WkT,  bk, nullptr, kf_bf, nullptr, 0, M_HIST, E_DIM, E_DIM, GM_BF16, stream);
    launch_gemm(hmk_bf,WvT,  bv, nullptr, vf_bf, nullptr, 0, M_HIST, E_DIM, E_DIM, GM_BF16, stream);
    launch_gemm(q_bf,  WqiT, ipb,           q2_f, nullptr, nullptr, 0, NTOK,   E_DIM, E_DIM, GM_F32, stream);
    launch_gemm(kf_bf, WkiT, ipb + E_DIM,   k2_f, nullptr, nullptr, 0, M_HIST, E_DIM, E_DIM, GM_F32, stream);
    launch_gemm(vf_bf, WviT, ipb + 2*E_DIM, v2_f, nullptr, nullptr, 0, M_HIST, E_DIM, E_DIM, GM_F32, stream);

    // ---- stage 2: attention + output proj + LN/gating ----
    attention_kernel<<<NTOK, 256, 0, stream>>>(q2_f, k2_f, v2_f, ctx_bf);
    launch_gemm(ctx_bf, aowT, aob, ao_f, nullptr, nullptr, 0, NTOK, E_DIM, E_DIM, GM_F32, stream);
    ln_gating_kernel<<<NTOK, 256, 0, stream>>>(ao_f, x, ln_g, ln_b, Wo, bo,
                                               Wg, bg, gates, out_logp, out_w,
                                               out_topi);

    // ---- stage 3: experts (dominant WMMA work) ----
    zero_kernel<<<512, 256, 0, stream>>>(out_final, NTOK * OUT_DIM);
    for (int e = 0; e < NE_; ++e) {
        launch_gemm(x_bf, W1T + (size_t)e * HID_DIM * E_DIM, b1 + (size_t)e * HID_DIM,
                    nullptr, h_bf, nullptr, 0, NTOK, HID_DIM, E_DIM, GM_GELU_BF16, stream);
        launch_gemm(h_bf, W2T + (size_t)e * OUT_DIM * HID_DIM, b2 + (size_t)e * OUT_DIM,
                    out_final, nullptr, gates + e, NE_, NTOK, OUT_DIM, HID_DIM,
                    GM_MOE_ACC, stream);
    }
}
